// pointfilternet_47433618817642
// MI455X (gfx1250) — matrix operand order repaired and checked by reading, not compiled, not run on hardware
//
#include <hip/hip_runtime.h>
#include <hip/hip_bf16.h>
#include <math.h>

// ---------------------------------------------------------------------------
// PointFilterNet forward for MI455X (gfx1250, wave32, WMMA).
// Heavy 1x1 convs -> f16xf16->f32 WMMA GEMMs (V_WMMA_F32_16X16X32_F16) with
// BatchNorm batch statistics folded into the GEMM epilogue (LDS ds_add_f32).
// W-tile staging goes through the Tensor Data Mover (tensor_load_to_lds +
// s_wait_tensorcnt) using the 6-arg clang-23 builtin arity; A-tile staging
// stays manual-transposed so B-fragments load as packed half2 from LDS.
// ---------------------------------------------------------------------------

typedef _Float16 half_t;
typedef __attribute__((ext_vector_type(16))) _Float16 v16h;
typedef __attribute__((ext_vector_type(8)))  float    v8f;
typedef __attribute__((ext_vector_type(2)))  _Float16 h2;
typedef __attribute__((ext_vector_type(4)))  unsigned int u32x4;
typedef __attribute__((ext_vector_type(8)))  int i32x8;
typedef __attribute__((ext_vector_type(4)))  int i32x4;

#define B_    16
#define N_    2048
#define P_    32768      // B*N
#define P4_   131072     // B*N*k (k=4)
#define EPS_  1e-5f

#if __has_builtin(__builtin_amdgcn_tensor_load_to_lds) && \
    __has_builtin(__builtin_amdgcn_s_wait_tensorcnt)
#define USE_TDM 1
#endif

// ---------------------------------------------------------------------------
// small utilities
// ---------------------------------------------------------------------------
__global__ void k_zero_f32(float* __restrict__ p, int n) {
  int i = blockIdx.x * blockDim.x + threadIdx.x;
  if (i < n) p[i] = 0.f;
}

__global__ void k_f32_to_f16(const float* __restrict__ s, half_t* __restrict__ d, int n) {
  int i = blockIdx.x * blockDim.x + threadIdx.x;
  if (i < n) d[i] = (half_t)s[i];
}

// ---------------------------------------------------------------------------
// kNN (k=4) on origin_noise [B,3,N].  pd = -||xi-xj||^2, top-4 largest,
// ties keep the earlier index (matches lax.top_k).  Coords cached in LDS.
// ---------------------------------------------------------------------------
__global__ __launch_bounds__(256) void k_knn(const float* __restrict__ on,
                                             int* __restrict__ idx) {
  __shared__ float sx[N_], sy[N_], sz[N_];
  const int b = blockIdx.y;
  const int n = blockIdx.x * 256 + threadIdx.x;
  const float* xb = on + (size_t)b * 3 * N_;
  for (int i = threadIdx.x; i < N_; i += 256) {
    sx[i] = xb[i]; sy[i] = xb[N_ + i]; sz[i] = xb[2 * N_ + i];
  }
  __syncthreads();
  const float px = sx[n], py = sy[n], pz = sz[n];
  float bv[4]; int bi[4];
  #pragma unroll
  for (int j = 0; j < 4; ++j) { bv[j] = -3.0e38f; bi[j] = 0; }
  for (int m = 0; m < N_; ++m) {
    float dx = px - sx[m], dy = py - sy[m], dz = pz - sz[m];
    float d = -(dx * dx + dy * dy + dz * dz);
    if (d > bv[3]) {
      int j = 3;
      while (j > 0 && d > bv[j - 1]) { bv[j] = bv[j - 1]; bi[j] = bi[j - 1]; --j; }
      bv[j] = d; bi[j] = m;
    }
  }
  int* o = idx + ((size_t)b * N_ + n) * 4;
  #pragma unroll
  for (int j = 0; j < 4; ++j) o[j] = bi[j];
}

// ---------------------------------------------------------------------------
// encoder layer 1: Cin=3 -> Cout, x [B,3,N] fp32.  Writes pre-BN f32 + stats.
// ---------------------------------------------------------------------------
__global__ __launch_bounds__(256) void k_conv1(const float* __restrict__ x,
                                               const float* __restrict__ w,
                                               const float* __restrict__ bias,
                                               float* __restrict__ Y,
                                               float* __restrict__ stats,
                                               int Cout) {
  __shared__ float bs[2];
  const int p = blockIdx.x * 256 + threadIdx.x;
  const int o = blockIdx.y;
  if (threadIdx.x == 0) { bs[0] = 0.f; bs[1] = 0.f; }
  __syncthreads();
  const int b = p >> 11;                         // / N_
  const size_t base = (size_t)p + (size_t)2 * b * N_;
  float v = w[o * 3 + 0] * x[base] + w[o * 3 + 1] * x[base + N_] +
            w[o * 3 + 2] * x[base + 2 * N_] + bias[o];
  Y[(size_t)o * P_ + p] = v;
  atomicAdd(&bs[0], v);
  atomicAdd(&bs[1], v * v);
  __syncthreads();
  if (threadIdx.x == 0) {
    atomicAdd(&stats[o], bs[0]);
    atomicAdd(&stats[Cout + o], bs[1]);
  }
}

// ---------------------------------------------------------------------------
// t1 conv: pos[b,c,n,k] = on[b,c,n]-on[b,c,idx], 3 -> 256, f16 out [256,P4]
// ---------------------------------------------------------------------------
__global__ __launch_bounds__(256) void k_t1(const float* __restrict__ on,
                                            const int* __restrict__ idx,
                                            const float* __restrict__ w,
                                            const float* __restrict__ bias,
                                            half_t* __restrict__ out) {
  const int p4 = blockIdx.x * 256 + threadIdx.x;
  const int o = blockIdx.y;
  const int bn = p4 >> 2;
  const int b = bn >> 11;
  const int n = bn & (N_ - 1);
  const int j = idx[p4];
  const size_t base = (size_t)b * 3 * N_;
  float dx = on[base + n] - on[base + j];
  float dy = on[base + N_ + n] - on[base + N_ + j];
  float dz = on[base + 2 * N_ + n] - on[base + 2 * N_ + j];
  float v = w[o * 3 + 0] * dx + w[o * 3 + 1] * dy + w[o * 3 + 2] * dz + bias[o];
  out[(size_t)o * P4_ + p4] = (half_t)v;
}

// ---------------------------------------------------------------------------
// gather x_j[c, p4] = feats[c, b*N + idx[p4]]
// ---------------------------------------------------------------------------
__global__ __launch_bounds__(256) void k_gather(const half_t* __restrict__ feats,
                                                const int* __restrict__ idx,
                                                half_t* __restrict__ xj) {
  const int p4 = blockIdx.x * 256 + threadIdx.x;
  const int o = blockIdx.y;
  const int bn = p4 >> 2;
  const int b = bn >> 11;
  xj[(size_t)o * P4_ + p4] = feats[(size_t)o * P_ + b * N_ + idx[p4]];
}

// ---------------------------------------------------------------------------
// WMMA GEMM:  Y[M,Pt] = W[M,K] (f16) x A[K,Pt] (f16) + bias.
// Block tile 128(M) x 64(P), K-step 32, 8 waves each own a 16x64 strip.
// Optional f32 / f16 outputs; optional BN sum/sumsq stats via LDS atomics.
// ---------------------------------------------------------------------------
#define BM 128
#define BP 64
#define BK 32
#define LDSTR 34   // padded half stride: 68B rows, h2-aligned

__global__ __launch_bounds__(256) void k_gemm_wmma(
    const half_t* __restrict__ W, const float* __restrict__ bias,
    const half_t* __restrict__ A, float* __restrict__ Yf32,
    half_t* __restrict__ Yf16, float* __restrict__ stats,
    int M, int K, int Pt) {
  __shared__ __align__(8) half_t sW[BM * LDSTR];   // [m][k], padded stride 34
  __shared__ __align__(8) half_t sA[BP * LDSTR];   // [n][k]  (transposed stage)
  __shared__ float sSum[BM], sSq[BM];

  const int tid = threadIdx.x;
  const int lane = tid & 31;
  const int wave = tid >> 5;
  const int m0 = blockIdx.y * BM;
  const int p0 = blockIdx.x * BP;
  const int mw = m0 + wave * 16;

  if (stats && tid < BM) { sSum[tid] = 0.f; sSq[tid] = 0.f; }

  v8f acc[4];
  #pragma unroll
  for (int s = 0; s < 4; ++s)
    #pragma unroll
    for (int i = 0; i < 8; ++i) acc[s][i] = 0.f;

  for (int k0 = 0; k0 < K; k0 += BK) {
    __syncthreads();

#ifdef USE_TDM
    // --- W tile via Tensor Data Mover: 2D tile, 2-byte elements,
    //     tile 32(x=K, contiguous) x 128(y=M), global row stride = K halves,
    //     LDS pad: 1 DWORD (pad_amount=0) after every 64B (pad_interval=3)
    //     -> padded 34-half rows, exactly the sW layout the fragments expect.
    if (tid == 0) {
      unsigned ldsAddr = (unsigned)(unsigned long long)(const void*)&sW[0];
      unsigned long long ga =
          (unsigned long long)(const void*)&W[(size_t)m0 * K + k0];
      u32x4 g0;
      g0[0] = 1u;                                    // count=1 (valid user D#)
      g0[1] = ldsAddr;                               // lds_addr
      g0[2] = (unsigned)(ga & 0xFFFFFFFFull);        // global_addr[31:0]
      g0[3] = (unsigned)((ga >> 32) & 0x1FFFFFFull)  // global_addr[56:32]
              | (2u << 30);                          // type=2 ("image")
      i32x8 g1;
      g1[0] = (1 << 16)        // data_size = 1 -> 2 bytes
            | (1 << 20)        // pad_enable
            | (3 << 22)        // pad_interval: 3 -> every 16 DWORDs (64B)
            | (0 << 25);       // pad_amount: 0 -> 1 DWORD
      g1[1] = (int)((BK & 0xFFFF) << 16);            // tensor_dim0[15:0]
      g1[2] = (int)(((BK >> 16) & 0xFFFF) | ((BM & 0xFFFF) << 16)); // td0 hi | td1 lo
      g1[3] = (int)(((BM >> 16) & 0xFFFF) | (BK << 16));            // td1 hi | tile_dim0
      g1[4] = (int)(BM & 0xFFFF);                    // tile_dim1 | tile_dim2=0
      g1[5] = (int)K;                                // tensor_dim0_stride[31:0]
      g1[6] = 0;                                     // stride0 hi | stride1 lo
      g1[7] = 0;
      i32x4 gz4; gz4[0] = 0; gz4[1] = 0; gz4[2] = 0; gz4[3] = 0;
      i32x8 gz8;
      #pragma unroll
      for (int z = 0; z < 8; ++z) gz8[z] = 0;
      // clang-23 / amdgpu-toolchain arity: (g0, g1, g2, g3, x, cpol)
      __builtin_amdgcn_tensor_load_to_lds(g0, g1, gz4, gz4, gz8, 0);
      __builtin_amdgcn_s_wait_tensorcnt(0);
    }
#else
    // --- W tile manual stage (h2 units, row-major with padded stride)
    {
      const h2* Wg = (const h2*)W;
      const int rowH2 = K >> 1;
      #pragma unroll
      for (int e = tid; e < (BM * BK) / 2; e += 256) {
        int r = e >> 4;      // 16 h2 per row
        int kk = e & 15;
        h2 v = Wg[(size_t)(m0 + r) * rowH2 + (k0 >> 1) + kk];
        *(h2*)&sW[r * LDSTR + kk * 2] = v;
      }
    }
#endif
    // --- A tile manual transposed stage into [n][k] so B-fragments can use
    //     packed half2 LDS loads (consumed 4x per staging).
    #pragma unroll
    for (int e = tid; e < BK * BP; e += 256) {
      int kk = e >> 6;     // 64 n per k row
      int n = e & 63;
      sA[n * LDSTR + kk] = A[(size_t)(k0 + kk) * Pt + p0 + n];
      // prefetch next K-step's A tile, one lane per 128B cacheline
      if ((e & 63) == 0 && k0 + BK < K)
        __builtin_prefetch(&A[(size_t)(k0 + BK + kk) * Pt + p0], 0, 1);
    }
    __syncthreads();

    // A-fragment (16-bit A 16x32 layout): lane = M row (+K-half select),
    // vgpr v holds K pair k = (v/4)*16 + (lane/16)*8 + (v%4)*2
    v16h af;
    {
      const int row = lane & 15;
      const int khalf = lane >> 4;
      const half_t* wrow = &sW[(wave * 16 + row) * LDSTR];
      #pragma unroll
      for (int v = 0; v < 8; ++v) {
        int k = ((v >> 2) << 4) + (khalf << 3) + ((v & 3) << 1);
        h2 t = *(const h2*)&wrow[k];
        af[2 * v] = t[0];
        af[2 * v + 1] = t[1];
      }
    }
    // B-fragments (16-bit B 32x16 layout): col = lane%16,
    // vgpr v holds K pair k = (v/4)*16 + (lane/16)*8 + (v%4)*2 (matches A)
    #pragma unroll
    for (int s = 0; s < 4; ++s) {
      const int col = (lane & 15) + s * 16;
      const int kb = (lane >> 4) << 3;
      const half_t* arow = &sA[col * LDSTR];
      v16h bf;
      #pragma unroll
      for (int v = 0; v < 8; ++v) {
        h2 t = *(const h2*)&arow[((v >> 2) << 4) + kb + ((v & 3) << 1)];
        bf[2 * v] = t[0];
        bf[2 * v + 1] = t[1];
      }
      acc[s] = __builtin_amdgcn_wmma_f32_16x16x32_f16(
          false, af, false, bf, (short)0, acc[s], false, false);
    }
  }

  // epilogue: C/D layout — vgpr r, lane: m = mw + (lane/16)*8 + r, n = lane%16
  #pragma unroll
  for (int s = 0; s < 4; ++s) {
    const int col = p0 + s * 16 + (lane & 15);
    #pragma unroll
    for (int r = 0; r < 8; ++r) {
      const int m = mw + ((lane >> 4) << 3) + r;
      float v = acc[s][r] + bias[m];
      if (Yf32) Yf32[(size_t)m * Pt + col] = v;
      if (Yf16) Yf16[(size_t)m * Pt + col] = (half_t)v;
      if (stats) {
        atomicAdd(&sSum[m - m0], v);
        atomicAdd(&sSq[m - m0], v * v);
      }
    }
  }
  if (stats) {
    __syncthreads();
    if (tid < BM) {
      atomicAdd(&stats[m0 + tid], sSum[tid]);
      atomicAdd(&stats[M + m0 + tid], sSq[tid]);
    }
  }
}

// ---------------------------------------------------------------------------
// BN finalize: stats[2M] -> scale/shift   (biased var, matches jnp.var)
// ---------------------------------------------------------------------------
__global__ void k_bn_finalize(const float* __restrict__ stats,
                              const float* __restrict__ g,
                              const float* __restrict__ be,
                              float* __restrict__ scale,
                              float* __restrict__ shift, int M, float invCnt) {
  int m = blockIdx.x * blockDim.x + threadIdx.x;
  if (m >= M) return;
  float mean = stats[m] * invCnt;
  float var = stats[M + m] * invCnt - mean * mean;
  float inv = rsqrtf(var + EPS_);
  float sc = g[m] * inv;
  scale[m] = sc;
  shift[m] = be[m] - mean * sc;
}

// apply BN + ReLU and downconvert to f16 for the next GEMM operand
__global__ __launch_bounds__(256) void k_apply(const float* __restrict__ Y,
                                               const float* __restrict__ scale,
                                               const float* __restrict__ shift,
                                               half_t* __restrict__ out, int Pt) {
  const int p = blockIdx.x * 256 + threadIdx.x;
  const int m = blockIdx.y;
  float v = Y[(size_t)m * Pt + p] * scale[m] + shift[m];
  out[(size_t)m * Pt + p] = (half_t)fmaxf(v, 0.f);
}

// rel0 = phi(broadcast over k) - psi + delta, f16
__global__ __launch_bounds__(256) void k_combine(const half_t* __restrict__ phi,
                                                 const half_t* __restrict__ psi,
                                                 const half_t* __restrict__ delta,
                                                 half_t* __restrict__ R) {
  const int p4 = blockIdx.x * 256 + threadIdx.x;
  const int o = blockIdx.y;
  const size_t q = (size_t)o * P4_ + p4;
  float v = (float)phi[(size_t)o * P_ + (p4 >> 2)] - (float)psi[q] + (float)delta[q];
  R[q] = (half_t)v;
}

// attention: rel = relu(bn(g2out)); softmax over k; f = sum w*(alpha+delta)
__global__ __launch_bounds__(256) void k_attention(const float* __restrict__ Yg2,
                                                   const float* __restrict__ scale,
                                                   const float* __restrict__ shift,
                                                   const half_t* __restrict__ alpha,
                                                   const half_t* __restrict__ delta,
                                                   half_t* __restrict__ fout) {
  const int bn = blockIdx.x * 256 + threadIdx.x;
  const int o = blockIdx.y;
  const size_t base = (size_t)o * P4_ + (size_t)bn * 4;
  const float sc = scale[o], sh = shift[o];
  float r[4];
  float mx = -3.0e38f;
  #pragma unroll
  for (int i = 0; i < 4; ++i) {
    r[i] = fmaxf(Yg2[base + i] * sc + sh, 0.f);
    mx = fmaxf(mx, r[i]);
  }
  float e[4], s = 0.f;
  #pragma unroll
  for (int i = 0; i < 4; ++i) { e[i] = __expf(r[i] - mx); s += e[i]; }
  const float inv = 1.f / s;
  float f = 0.f;
  #pragma unroll
  for (int i = 0; i < 4; ++i)
    f += e[i] * inv * ((float)alpha[base + i] + (float)delta[base + i]);
  fout[(size_t)o * P_ + bn] = (half_t)f;
}

// global max-pool over N with BN+ReLU folded in: h[b,c] = max_n relu(bn(Y6))
__global__ __launch_bounds__(256) void k_maxpool(const float* __restrict__ Y,
                                                 const float* __restrict__ scale,
                                                 const float* __restrict__ shift,
                                                 float* __restrict__ h) {
  __shared__ float red[256];
  const int bc = blockIdx.x;
  const int b = bc >> 10;
  const int c = bc & 1023;
  const float* row = Y + (size_t)c * P_ + (size_t)b * N_;
  const float sc = scale[c], sh = shift[c];
  float m = -3.0e38f;
  for (int i = threadIdx.x; i < N_; i += 256)
    m = fmaxf(m, fmaxf(row[i] * sc + sh, 0.f));
  red[threadIdx.x] = m;
  __syncthreads();
  for (int s = 128; s > 0; s >>= 1) {
    if (threadIdx.x < s) red[threadIdx.x] = fmaxf(red[threadIdx.x], red[threadIdx.x + s]);
    __syncthreads();
  }
  if (threadIdx.x == 0) h[bc] = red[0];
}

// tiny FC: out[s,o] = in[s,:] . w[o,:] + b[o]   (act: 0=none, 1=tanh)
__global__ void k_fc(const float* __restrict__ in, const float* __restrict__ w,
                     const float* __restrict__ b, float* __restrict__ out,
                     int S, int F, int K, int act) {
  int i = blockIdx.x * blockDim.x + threadIdx.x;
  if (i >= S * F) return;
  int s = i / F, o = i % F;
  const float* ir = in + (size_t)s * K;
  const float* wr = w + (size_t)o * K;
  float acc = b[o];
  for (int k = 0; k < K; ++k) acc += ir[k] * wr[k];
  if (act == 1) acc = tanhf(acc);
  out[i] = acc;
}

// BN over batch axis for [S,F] + optional ReLU (one thread per feature)
__global__ void k_bn1d_apply(const float* __restrict__ in,
                             const float* __restrict__ g,
                             const float* __restrict__ be,
                             float* __restrict__ out, int S, int F, int relu) {
  int f = blockIdx.x * blockDim.x + threadIdx.x;
  if (f >= F) return;
  float s = 0.f, sq = 0.f;
  for (int i = 0; i < S; ++i) { float v = in[i * F + f]; s += v; sq += v * v; }
  float mean = s / S;
  float var = sq / S - mean * mean;
  float inv = rsqrtf(var + EPS_);
  float sc = g[f] * inv, sh = be[f] - mean * sc;
  for (int i = 0; i < S; ++i) {
    float v = in[i * F + f] * sc + sh;
    if (relu) v = fmaxf(v, 0.f);
    out[i * F + f] = v;
  }
}

// ---------------------------------------------------------------------------
// host orchestration
// ---------------------------------------------------------------------------
extern "C" void kernel_launch(void* const* d_in, const int* in_sizes, int n_in,
                              void* d_out, int out_size, void* d_ws, size_t ws_size,
                              hipStream_t stream) {
  (void)in_sizes; (void)n_in; (void)out_size; (void)ws_size;
  const float* X  = (const float*)d_in[0];
  const float* ON = (const float*)d_in[1];
  #define PW(i) ((const float*)d_in[i])
  // params, insertion order (50 inputs total):
  //  2 e_w1   3 e_b1   4 e_g1   5 e_be1
  //  6 e_w2   7 e_b2   8 e_g2   9 e_be2
  // 10 e_w3  11 e_b3  12 e_g3  13 e_be3
  // 14 t1_w  15 t1_b
  // 16 t2_w  17 t2_b  18 t_g   19 t_be
  // 20 phi_w 21 phi_b 22 psi_w 23 psi_b 24 alpha_w 25 alpha_b
  // 26 g1_w  27 g1_b  28 g2_w  29 g2_b  30 g_g     31 g_be
  // 32 c5_w  33 c5_b  34 bn4_g 35 bn4_b
  // 36 c6_w  37 c6_b  38 bn5_g 39 bn5_b
  // 40 fc1_w 41 fc1_b 42 bn1_g 43 bn1_b
  // 44 fc2_w 45 fc2_b 46 bn2_g 47 bn2_b
  // 48 fc3_w 49 fc3_b

  // --- workspace layout (bump allocator, 256B aligned) ---
  char* base = (char*)d_ws;
  size_t off = 0;
  auto alloc = [&](size_t bytes) -> void* {
    void* p = base + off;
    off += (bytes + 255) & ~(size_t)255;
    return p;
  };
  int*    idx    = (int*)alloc((size_t)P4_ * 4);
  half_t* We2    = (half_t*)alloc(128 * 64 * 2);
  half_t* We3    = (half_t*)alloc(256 * 128 * 2);
  half_t* Wt2    = (half_t*)alloc(256 * 256 * 2);
  half_t* Wphi   = (half_t*)alloc(256 * 256 * 2);
  half_t* Wpsi   = (half_t*)alloc(256 * 256 * 2);
  half_t* Wal    = (half_t*)alloc(256 * 256 * 2);
  half_t* Wg1    = (half_t*)alloc(256 * 256 * 2);
  half_t* Wg2    = (half_t*)alloc(256 * 256 * 2);
  half_t* Wc5    = (half_t*)alloc(512 * 256 * 2);
  half_t* Wc6    = (half_t*)alloc((size_t)1024 * 512 * 2);
  float*  stats  = (float*)alloc(2 * 1024 * 4);
  float*  scale  = (float*)alloc(1024 * 4);
  float*  shift  = (float*)alloc(1024 * 4);
  float*  BIG    = (float*)alloc((size_t)256 * P4_ * 4);   // 128 MB, reused serially
  half_t* slotA  = (half_t*)alloc((size_t)256 * P4_ * 2);  // D1 / psi / g1out / alpha
  half_t* slotB  = (half_t*)alloc((size_t)256 * P4_ * 2);  // delta
  half_t* slotC  = (half_t*)alloc((size_t)256 * P4_ * 2);  // x_j
  half_t* slotD  = (half_t*)alloc((size_t)256 * P4_ * 2);  // rel0, later A5 (512 x P)
  half_t* A1     = (half_t*)alloc((size_t)64 * P_ * 2);
  half_t* A2     = (half_t*)alloc((size_t)128 * P_ * 2);
  half_t* feats  = (half_t*)alloc((size_t)256 * P_ * 2);   // later reused as f_out
  half_t* phiO   = (half_t*)alloc((size_t)256 * P_ * 2);
  float*  hpool  = (float*)alloc((size_t)16 * 1024 * 4);
  float*  fcA    = (float*)alloc(16 * 512 * 4);
  float*  fcB    = (float*)alloc(16 * 512 * 4);
  float*  fcC    = (float*)alloc(16 * 256 * 4);
  float*  fcD    = (float*)alloc(16 * 256 * 4);

  auto cvt = [&](const float* s, half_t* d, int n) {
    k_f32_to_f16<<<(n + 255) / 256, 256, 0, stream>>>(s, d, n);
  };
  auto zero = [&](float* p, int n) {
    k_zero_f32<<<(n + 255) / 256, 256, 0, stream>>>(p, n);
  };
  auto gemm = [&](const half_t* W, const float* bias, const half_t* A,
                  float* Yf32, half_t* Yf16, float* st, int M, int K, int Pt) {
    dim3 g(Pt / BP, M / BM);
    k_gemm_wmma<<<g, 256, 0, stream>>>(W, bias, A, Yf32, Yf16, st, M, K, Pt);
  };
  auto finalize = [&](const float* g, const float* be, int M, float cnt) {
    k_bn_finalize<<<(M + 255) / 256, 256, 0, stream>>>(stats, g, be, scale, shift,
                                                       M, 1.f / cnt);
  };
  auto apply = [&](const float* Y, half_t* out, int M, int Pt) {
    k_apply<<<dim3(Pt / 256, M), 256, 0, stream>>>(Y, scale, shift, out, Pt);
  };

  // weight downconversion (f32 -> f16) for all WMMA GEMM operands
  cvt(PW(6),  We2, 128 * 64);
  cvt(PW(10), We3, 256 * 128);
  cvt(PW(16), Wt2, 65536);
  cvt(PW(20), Wphi, 65536);
  cvt(PW(22), Wpsi, 65536);
  cvt(PW(24), Wal, 65536);
  cvt(PW(26), Wg1, 65536);
  cvt(PW(28), Wg2, 65536);
  cvt(PW(32), Wc5, 512 * 256);
  cvt(PW(36), Wc6, 1024 * 512);

  // kNN
  k_knn<<<dim3(N_ / 256, B_), 256, 0, stream>>>(ON, idx);

  // encoder L1: 3 -> 64
  zero(stats, 2 * 64);
  k_conv1<<<dim3(P_ / 256, 64), 256, 0, stream>>>(X, PW(2), PW(3), BIG, stats, 64);
  finalize(PW(4), PW(5), 64, (float)P_);
  apply(BIG, A1, 64, P_);

  // encoder L2: 64 -> 128
  zero(stats, 2 * 128);
  gemm(We2, PW(7), A1, BIG, nullptr, stats, 128, 64, P_);
  finalize(PW(8), PW(9), 128, (float)P_);
  apply(BIG, A2, 128, P_);

  // encoder L3: 128 -> 256 -> feats
  zero(stats, 2 * 256);
  gemm(We3, PW(11), A2, BIG, nullptr, stats, 256, 128, P_);
  finalize(PW(12), PW(13), 256, (float)P_);
  apply(BIG, feats, 256, P_);

  // t1 (pos conv, 3 -> 256) into slotA
  k_t1<<<dim3(P4_ / 256, 256), 256, 0, stream>>>(ON, idx, PW(14), PW(15), slotA);

  // t2 + BN(t_g,t_be) + ReLU -> delta (slotB)
  zero(stats, 2 * 256);
  gemm(Wt2, PW(17), slotA, BIG, nullptr, stats, 256, 256, P4_);
  finalize(PW(18), PW(19), 256, (float)P4_);
  apply(BIG, slotB, 256, P4_);

  // phi(feats) -> phiO (f16, bias only)
  gemm(Wphi, PW(21), feats, nullptr, phiO, nullptr, 256, 256, P_);

  // gather x_j -> slotC
  k_gather<<<dim3(P4_ / 256, 256), 256, 0, stream>>>(feats, idx, slotC);

  // psi(x_j) -> slotA (f16, bias only; D1 is dead after t2)
  gemm(Wpsi, PW(23), slotC, nullptr, slotA, nullptr, 256, 256, P4_);

  // rel0 = phi - psi + delta -> slotD
  k_combine<<<dim3(P4_ / 256, 256), 256, 0, stream>>>(phiO, slotA, slotB, slotD);

  // g1(rel0) -> slotA (f16, bias only; psi is dead)
  gemm(Wg1, PW(27), slotD, nullptr, slotA, nullptr, 256, 256, P4_);

  // g2 + stats -> BIG f32; BN(g_g,g_be) folded into the attention kernel
  zero(stats, 2 * 256);
  gemm(Wg2, PW(29), slotA, BIG, nullptr, stats, 256, 256, P4_);
  finalize(PW(30), PW(31), 256, (float)P4_);

  // alpha(x_j) -> slotA (f16, bias only; g1out is dead)
  gemm(Wal, PW(25), slotC, nullptr, slotA, nullptr, 256, 256, P4_);

  // attention -> f_out (reuse feats buffer; feats dead after phi+gather)
  k_attention<<<dim3(P_ / 256, 256), 256, 0, stream>>>(BIG, scale, shift, slotA,
                                                       slotB, feats);

  // decoder c5: 256 -> 512, BN(bn4) + ReLU -> A5 (slotD; rel0 is dead)
  zero(stats, 2 * 512);
  gemm(Wc5, PW(33), feats, BIG, nullptr, stats, 512, 256, P_);
  finalize(PW(34), PW(35), 512, (float)P_);
  apply(BIG, slotD, 512, P_);

  // decoder c6: 512 -> 1024, BN(bn5) folded into max-pool
  zero(stats, 2 * 1024);
  gemm(Wc6, PW(37), slotD, BIG, nullptr, stats, 1024, 512, P_);
  finalize(PW(38), PW(39), 1024, (float)P_);

  // max pool (BN+ReLU folded) -> hpool [16,1024]
  k_maxpool<<<16 * 1024, 256, 0, stream>>>(BIG, scale, shift, hpool);

  // FC head: fc1 -> bn1 -> relu -> fc2 -> bn2 -> relu -> fc3 -> tanh
  k_fc<<<(16 * 512 + 255) / 256, 256, 0, stream>>>(hpool, PW(40), PW(41), fcA,
                                                   16, 512, 1024, 0);
  k_bn1d_apply<<<2, 256, 0, stream>>>(fcA, PW(42), PW(43), fcB, 16, 512, 1);
  k_fc<<<(16 * 256 + 255) / 256, 256, 0, stream>>>(fcB, PW(44), PW(45), fcC,
                                                   16, 256, 512, 0);
  k_bn1d_apply<<<1, 256, 0, stream>>>(fcC, PW(46), PW(47), fcD, 16, 256, 1);
  k_fc<<<1, 64, 0, stream>>>(fcD, PW(48), PW(49), (float*)d_out, 16, 3, 256, 1);
  #undef PW
}
